// CumulHazardFunctionNetwork_59992103190734
// MI455X (gfx1250) — matrix operand order repaired and checked by reading, not compiled, not run on hardware
//
#include <hip/hip_runtime.h>

typedef _Float16 h16;
typedef __attribute__((ext_vector_type(4)))  _Float16 v4h;
typedef __attribute__((ext_vector_type(8)))  _Float16 v8h;
typedef __attribute__((ext_vector_type(16))) _Float16 v16h;
typedef __attribute__((ext_vector_type(8)))  float    v8f;

#define WMMA_F16(A, B, C) \
  __builtin_amdgcn_wmma_f32_16x16x32_f16(false, (A), false, (B), (short)0, (C), false, false)

// Branchless tanh: single V_TANH_F32 if available, else exp/rcp form that
// saturates correctly (e=inf -> 1, e=0 -> -1) with no divergent branches.
__device__ __forceinline__ float fast_tanh(float x) {
#if __has_builtin(__builtin_amdgcn_tanh_f32)
  return __builtin_amdgcn_tanh_f32(x);
#else
  float e = __expf(2.f * x);
  return 1.f - 2.f / (e + 1.f);
#endif
}

// A fragment (16x32 f16, row-major source, stride 256 halves):
// lane<16: row=lane, K = {k0..k0+7, k0+16..k0+23}; lane>=16: row=lane-16, K shifted by 8.
// Caller passes p = base + (m0 + (lane&15))*256 + k0 + (lane>>4)*8.
__device__ __forceinline__ v16h ldfragA(const h16* p) {
  v8h lo = *(const v8h*)p;
  v8h hi = *(const v8h*)(p + 16);
  return __builtin_shufflevector(lo, hi, 0,1,2,3,4,5,6,7,8,9,10,11,12,13,14,15);
}

// B fragment (32x16 f16, weights row-major [n][k], stride 256):
// lane<16: col N=lane, K = {k0..k0+7, k0+16..k0+23}; lane>=16: col N=lane-16, K shifted by 8.
// Caller passes p = W + (n0 + (lane&15))*256 + k0 + (lane>>4)*8.
__device__ __forceinline__ v16h ldfragB(const h16* p) {
  v8h lo = *(const v8h*)p;
  v8h hi = *(const v8h*)(p + 16);
  return __builtin_shufflevector(lo, hi, 0,1,2,3,4,5,6,7,8,9,10,11,12,13,14,15);
}

// Single-stream GEMM phase: [64x256] f16 LDS input vs [256x256] f16 global
// weight (row-major [n][k]). Wave owns columns [nb0, nb0+32).
template <typename EPI>
__device__ __forceinline__ void gemm_phase(const h16* IN, const h16* __restrict__ W,
                                           int r, int hi, int nb0, EPI epi) {
  v16h Bf0[8], Bf1[8];
  {
    const h16* wp0 = W + (nb0 + r) * 256 + hi * 8;
    const h16* wp1 = wp0 + 16 * 256;
#pragma unroll
    for (int k = 0; k < 8; ++k) {
      Bf0[k] = ldfragB(wp0 + k * 32);
      Bf1[k] = ldfragB(wp1 + k * 32);
    }
  }
#pragma unroll
  for (int m = 0; m < 4; ++m) {
    const h16* ap = IN + (m * 16 + r) * 256 + hi * 8;
    v8f a0 = {};
    v8f a1 = {};
#pragma unroll
    for (int k = 0; k < 8; ++k) {
      v16h Av = ldfragA(ap + k * 32);
      a0 = WMMA_F16(Av, Bf0[k], a0);
      a1 = WMMA_F16(Av, Bf1[k], a1);
    }
    epi(m, a0, a1);
  }
}

// Dual-stream GEMM phase: value stream INv and tangent stream INd share the
// same B fragments (one weight read, 4 WMMAs per K-step pair of A loads).
template <typename EPI>
__device__ __forceinline__ void gemm_dual(const h16* INv, const h16* INd,
                                          const h16* __restrict__ W,
                                          int r, int hi, int nb0, EPI epi) {
  v16h Bf0[8], Bf1[8];
  {
    const h16* wp0 = W + (nb0 + r) * 256 + hi * 8;
    const h16* wp1 = wp0 + 16 * 256;
#pragma unroll
    for (int k = 0; k < 8; ++k) {
      Bf0[k] = ldfragB(wp0 + k * 32);
      Bf1[k] = ldfragB(wp1 + k * 32);
    }
  }
#pragma unroll
  for (int m = 0; m < 4; ++m) {
    const h16* apv = INv + (m * 16 + r) * 256 + hi * 8;
    const h16* apd = INd + (m * 16 + r) * 256 + hi * 8;
    v8f a0 = {};
    v8f a1 = {};
    v8f d0 = {};
    v8f d1 = {};
#pragma unroll
    for (int k = 0; k < 8; ++k) {
      v16h Av = ldfragA(apv + k * 32);
      v16h Ad = ldfragA(apd + k * 32);
      a0 = WMMA_F16(Av, Bf0[k], a0);
      a1 = WMMA_F16(Av, Bf1[k], a1);
      d0 = WMMA_F16(Ad, Bf0[k], d0);
      d1 = WMMA_F16(Ad, Bf1[k], d1);
    }
    epi(m, a0, a1, d0, d1);
  }
}

// ---- prep: f16 weight conversion + rank-1 folding of the time branch ----
// v2[n] = sum_c w2[n][256+c] * w1[c]   (also the constant JVP tangent into tanh1)
// u2[n] = sum_c w2[n][256+c] * b1[c]
__global__ __launch_bounds__(256) void chfn_prep(
    const float* __restrict__ w1, const float* __restrict__ b1,
    const float* __restrict__ w2, const float* __restrict__ wA,
    const float* __restrict__ wB, const float* __restrict__ w3,
    h16* __restrict__ w2h, h16* __restrict__ wAh, h16* __restrict__ wBh,
    h16* __restrict__ w3h, float* __restrict__ v2, float* __restrict__ u2) {
  int idx = blockIdx.x * blockDim.x + threadIdx.x;
  int nt = gridDim.x * blockDim.x;
  for (int i = idx; i < 256 * 256; i += nt) {
    int n = i >> 8, k = i & 255;
    w2h[i] = (h16)w2[n * 512 + k];  // only first 256 input features (the h part)
    wAh[i] = (h16)wA[i];
    wBh[i] = (h16)wB[i];
  }
  for (int i = idx; i < 16 * 256; i += nt) {
    int n = i >> 8;
    w3h[i] = (n < 10) ? (h16)w3[i] : (h16)0.0f;  // pad output dim 10 -> 16
  }
  for (int i = idx; i < 256; i += nt) {
    float s = 0.f, u = 0.f;
    for (int c = 0; c < 256; ++c) {
      float w = w2[i * 512 + 256 + c];
      s += w * w1[c];
      u += w * b1[c];
    }
    v2[i] = s;
    u2[i] = u;
  }
}

// ---- fused forward + JVP kernel: 64 tokens per block, 8 wave32s ----
__global__ __launch_bounds__(256) void chfn_main(
    const float* __restrict__ hsrc, const float* __restrict__ td,
    const float* __restrict__ b2, const float* __restrict__ bA,
    const float* __restrict__ bB, const float* __restrict__ b3,
    const h16* __restrict__ w2h, const h16* __restrict__ wAh,
    const h16* __restrict__ wBh, const h16* __restrict__ w3h,
    const float* __restrict__ v2g, const float* __restrict__ u2g,
    float* __restrict__ out, int ntok) {
  __shared__ h16 S0[64 * 256];  // h, later out3
  __shared__ h16 S1[64 * 256];  // out2, later out4
  __shared__ h16 S2[64 * 256];  // dot2, later dot4
  __shared__ h16 S3[64 * 256];  // dot3

  const int tid = threadIdx.x;
  const int lane = tid & 31, wave = tid >> 5;
  const int r = lane & 15, hi = lane >> 4;
  const int tok0 = blockIdx.x * 64;
  const int nb0 = wave * 32;
  const int c0 = nb0 + r, c1 = c0 + 16;

  // Stage h -> LDS (f32 -> f16), vectorized float4 reads / b64 LDS writes.
  {
    const float4* hp = (const float4*)(hsrc + (size_t)tok0 * 256);
#pragma unroll 4
    for (int i = tid; i < 64 * 64; i += 256) {
      float4 v = hp[i];
      v4h o = {(h16)v.x, (h16)v.y, (h16)v.z, (h16)v.w};
      *(v4h*)(S0 + (i >> 6) * 256 + (i & 63) * 4) = o;
    }
  }
  __syncthreads();

  // ---- Layer 1: z2 = h @ w2[:, :256].T + td*v2 + u2 + b2 ----
  // out2 = tanh(z2) -> S1 ;  dot2 = (1 - out2^2) * v2 -> S2  (JVP is elementwise here)
  gemm_phase(S0, w2h, r, hi, nb0, [&](int m, v8f a0, v8f a1) {
    float v20 = v2g[c0], u0 = u2g[c0] + b2[c0];
    float v21 = v2g[c1], u1 = u2g[c1] + b2[c1];
    int base = (m * 16 + 8 * hi) * 256 + c0;  // j indexes rows: +j*256
    asm volatile("" : "+v"(base));  // opaque: force 1 base reg + imm-offset stores
    h16* p1 = S1 + base;
    h16* p2 = S2 + base;
    const float* tdp = td + tok0 + m * 16 + 8 * hi;
#pragma unroll
    for (int j = 0; j < 8; ++j) {
      float t = tdp[j];
      float o0 = fast_tanh(a0[j] + t * v20 + u0);
      float o1 = fast_tanh(a1[j] + t * v21 + u1);
      p1[j * 256] = (h16)o0;
      p1[j * 256 + 16] = (h16)o1;
      p2[j * 256] = (h16)((1.f - o0 * o0) * v20);
      p2[j * 256 + 16] = (h16)((1.f - o1 * o1) * v21);
    }
  });
  __syncthreads();

  // ---- Layer 2 (fused value+tangent, one pass over wA):
  //   out3 = tanh(out2 @ wA.T + bA)        -> S0
  //   dot3 = (1 - out3^2) * (dot2 @ wA.T)  -> S3
  gemm_dual(S1, S2, wAh, r, hi, nb0, [&](int m, v8f a0, v8f a1, v8f d0, v8f d1) {
    float bb0 = bA[c0], bb1 = bA[c1];
    int base = (m * 16 + 8 * hi) * 256 + c0;
    asm volatile("" : "+v"(base));
    h16* pv = S0 + base;
    h16* pd = S3 + base;
#pragma unroll
    for (int j = 0; j < 8; ++j) {
      float o0 = fast_tanh(a0[j] + bb0);
      float o1 = fast_tanh(a1[j] + bb1);
      pv[j * 256] = (h16)o0;
      pv[j * 256 + 16] = (h16)o1;
      pd[j * 256] = (h16)((1.f - o0 * o0) * d0[j]);
      pd[j * 256 + 16] = (h16)((1.f - o1 * o1) * d1[j]);
    }
  });
  __syncthreads();

  // ---- Layer 3 (fused value+tangent, one pass over wB):
  //   out4 = tanh(out3 @ wB.T + bB)        -> S1
  //   dot4 = (1 - out4^2) * (dot3 @ wB.T)  -> S2
  gemm_dual(S0, S3, wBh, r, hi, nb0, [&](int m, v8f a0, v8f a1, v8f d0, v8f d1) {
    float bb0 = bB[c0], bb1 = bB[c1];
    int base = (m * 16 + 8 * hi) * 256 + c0;
    asm volatile("" : "+v"(base));
    h16* pv = S1 + base;
    h16* pd = S2 + base;
#pragma unroll
    for (int j = 0; j < 8; ++j) {
      float o0 = fast_tanh(a0[j] + bb0);
      float o1 = fast_tanh(a1[j] + bb1);
      pv[j * 256] = (h16)o0;
      pv[j * 256 + 16] = (h16)o1;
      pd[j * 256] = (h16)((1.f - o0 * o0) * d0[j]);
      pd[j * 256 + 16] = (h16)((1.f - o1 * o1) * d1[j]);
    }
  });
  __syncthreads();

  // ---- Head: z5 = out4 @ w3.T + b3 (N padded 10->16), z5_dot = dot4 @ w3.T ----
  // integral = softplus(z5); derivative = sigmoid(z5) * z5_dot / ntok
  if (wave < 4) {
    const int m = wave;  // each of waves 0..3 owns one 16-row M-tile
    const h16* ap1 = S1 + (m * 16 + r) * 256 + hi * 8;
    const h16* ap2 = S2 + (m * 16 + r) * 256 + hi * 8;
    const h16* wp = w3h + r * 256 + hi * 8;
    v8f zi = {};
    v8f zd = {};
#pragma unroll
    for (int k = 0; k < 8; ++k) {
      v16h Bf = ldfragB(wp + k * 32);
      zi = WMMA_F16(ldfragA(ap1 + k * 32), Bf, zi);
      zd = WMMA_F16(ldfragA(ap2 + k * 32), Bf, zd);
    }
    if (r < 10) {
      float b3v = b3[r];
      float scale = 1.f / (float)ntok;
      float* oi = out + (size_t)(tok0 + m * 16 + 8 * hi) * 10 + r;
      float* od = oi + (size_t)ntok * 10;
#pragma unroll
      for (int j = 0; j < 8; ++j) {
        float z = zi[j] + b3v;
        // Branchless softplus / sigmoid via hardware exp/log + cndmask selects.
        float e = __expf(z);
        float sp = __logf(1.f + e);
        sp = (z > 15.f) ? z : sp;    // avoid inf from e, exact tail
        sp = (z < -15.f) ? e : sp;   // keep relative accuracy in lower tail
        float sg = e / (1.f + e);
        sg = (z > 15.f) ? 1.f : sg;
        oi[j * 10] = sp;
        od[j * 10] = sg * zd[j] * scale;
      }
    }
  }
}

extern "C" void kernel_launch(void* const* d_in, const int* in_sizes, int n_in,
                              void* d_out, int out_size, void* d_ws, size_t ws_size,
                              hipStream_t stream) {
  (void)n_in; (void)out_size; (void)ws_size;
  const float* hs = (const float*)d_in[0];
  const float* td = (const float*)d_in[1];
  const float* w1 = (const float*)d_in[2];
  const float* b1 = (const float*)d_in[3];
  const float* w2 = (const float*)d_in[4];
  const float* b2 = (const float*)d_in[5];
  const float* wA = (const float*)d_in[6];
  const float* bA = (const float*)d_in[7];
  const float* wB = (const float*)d_in[8];
  const float* bB = (const float*)d_in[9];
  const float* w3 = (const float*)d_in[10];
  const float* b3 = (const float*)d_in[11];

  char* ws = (char*)d_ws;
  h16* w2h = (h16*)(ws + 0);          // 256*256*2 = 131072 B
  h16* wAh = (h16*)(ws + 131072);     // 131072 B
  h16* wBh = (h16*)(ws + 262144);     // 131072 B
  h16* w3h = (h16*)(ws + 393216);     // 16*256*2 = 8192 B
  float* v2 = (float*)(ws + 401408);  // 1024 B
  float* u2 = (float*)(ws + 402432);  // 1024 B

  const int ntok = in_sizes[1];  // B*S = 131072

  chfn_prep<<<256, 256, 0, stream>>>(w1, b1, w2, wA, wB, w3, w2h, wAh, wBh, w3h, v2, u2);
  chfn_main<<<ntok / 64, 256, 0, stream>>>(hs, td, b2, bA, bB, b3, w2h, wAh, wBh, w3h,
                                           v2, u2, (float*)d_out, ntok);
}
